// RNN_83674552861233
// MI455X (gfx1250) — matrix operand order repaired and checked by reading, not compiled, not run on hardware
//
#include <hip/hip_runtime.h>
#include <hip/hip_bf16.h>

typedef __attribute__((ext_vector_type(16))) _Float16 v16h;
typedef __attribute__((ext_vector_type(8)))  _Float16 v8h;
typedef __attribute__((ext_vector_type(8)))  float    v8f;
typedef int v4i_ __attribute__((__vector_size__(16)));
typedef __attribute__((address_space(1))) v4i_ as1_v4i;   // global
typedef __attribute__((address_space(3))) v4i_ as3_v4i;   // LDS

#define B_      4096
#define T_      60
#define F_      128
#define H_      256
#define KZ      384          // F_ + H_
#define NG      1024         // 4 * H_
#define BT      64           // batch rows per group
#define THREADS 512          // 16 waves
#define ZP      392          // Zs pitch (halves): 384 + 8 -> conflict-free b128
#define WP      264          // Wlds pitch (halves): 256 + 8 -> conflict-free b128
#define GROUPS  64           // 4096 / 64
#define QWG     4            // WGs per group (hidden-column quarters)

// workspace layout (bytes)
#define WZ_OFF   0           // 384*1024 f16 = 786432
#define BIAS_OFF 786432      // 1024 f32
#define V_OFF    790528      // 256 f32
#define S_OFF    791552      // 1 f32
#define CNT_OFF  792064      // GROUPS ints (barrier counters)
#define HBUF_OFF 794624      // 2 * GROUPS * 64*256 f16 = 4 MB

__device__ __forceinline__ float fast_rcp(float x) {
    return __builtin_amdgcn_rcpf(x);         // v_rcp_f32, ~1 ulp
}
__device__ __forceinline__ float sigf(float x) {
    return fast_rcp(1.0f + __expf(-x));
}
__device__ __forceinline__ float tanhf_fast(float x) {
    float e = __expf(2.0f * x);
    return 1.0f - 2.0f * fast_rcp(e + 1.0f); // stable at +-inf
}
__device__ __forceinline__ v8f wmma_f16(v16h a, v16h b, v8f c) {
    return __builtin_amdgcn_wmma_f32_16x16x32_f16(false, a, false, b, (short)0, c, false, false);
}

// ---------------------------------------------------------------------------
// Prep 1: fuse weights -> f16 [384][1024], fuse biases, zero barrier counters.
// ---------------------------------------------------------------------------
__global__ void prep_weights(const float* __restrict__ Wf, const float* __restrict__ bf,
                             const float* __restrict__ Wu, const float* __restrict__ bu,
                             const float* __restrict__ Wc, const float* __restrict__ bc,
                             const float* __restrict__ Wo, const float* __restrict__ bo,
                             _Float16* __restrict__ Wz, float* __restrict__ bias,
                             int* __restrict__ cnt) {
    int idx = blockIdx.x * blockDim.x + threadIdx.x;
    if (idx < KZ * NG) {
        int r = idx >> 10, n = idx & 1023;
        int g = n >> 8, c = n & 255;
        const float* Ws = (g == 0) ? Wf : (g == 1) ? Wu : (g == 2) ? Wc : Wo;
        Wz[idx] = (_Float16)Ws[r * H_ + c];
    }
    if (idx < NG) {
        int g = idx >> 8, c = idx & 255;
        const float* bs = (g == 0) ? bf : (g == 1) ? bu : (g == 2) ? bc : bo;
        bias[idx] = bs[c];
    }
    if (idx < GROUPS) cnt[idx] = 0;
}

// ---------------------------------------------------------------------------
// Prep 2: fold head:  v[i] = sum_j Wd1[i][j]*Wd2[j];  s = bd1.Wd2 + bd2
// ---------------------------------------------------------------------------
__global__ void prep_head(const float* __restrict__ Wd1, const float* __restrict__ bd1,
                          const float* __restrict__ Wd2, const float* __restrict__ bd2,
                          float* __restrict__ v, float* __restrict__ s) {
    int i = threadIdx.x;
    float acc = 0.f;
    for (int j = 0; j < H_; ++j) acc += Wd1[i * H_ + j] * Wd2[j];
    v[i] = acc;
    if (i == 0) {
        float a = bd2[0];
        for (int j = 0; j < H_; ++j) a += bd1[j] * Wd2[j];
        *s = a;
    }
}

// ---------------------------------------------------------------------------
// Main: group of 4 WGs = one 64-row batch tile. WG q owns hidden cols
// [64q,64q+64) of all four gates: weight slice 384x256 f16 resident in LDS.
// Per step: each of 16 waves computes one (mtile, hsub) unit: 4 gate accs,
// 12 k-steps * 4 gates = 48 wmma; c stays in VGPRs; h exchanged via L2
// (async-to-LDS on the read side).
// ---------------------------------------------------------------------------
__global__ __launch_bounds__(THREADS) void lstm_main(
        const float* __restrict__ x, const _Float16* __restrict__ Wz,
        const float* __restrict__ bias, const float* __restrict__ vhead,
        const float* __restrict__ shead, _Float16* hbuf, int* cnt,
        float* __restrict__ out) {

    __shared__ _Float16 Zs[BT][ZP];       // [row][384] f16: [x_t | h_t]
    __shared__ _Float16 Wlds[KZ][WP];     // [k][n'] f16, n' = gate*64 + local col

    const int tid    = threadIdx.x;
    const int lane   = tid & 31;
    const int wave   = tid >> 5;          // 0..15
    const int mtile  = wave & 3;          // 16-row M tile
    const int hsub   = wave >> 2;         // 16-col tile within our 64 cols
    const int rowA   = lane & 15;
    const int khalf  = lane >> 4;
    const int group  = blockIdx.x >> 2;   // 0..63
    const int q      = blockIdx.x & 3;    // column quarter
    const int batch0 = group * BT;

    int* cnt_g = cnt + group;
    _Float16* hb0 = hbuf + (size_t)group * (BT * H_);            // t even
    _Float16* hb1 = hbuf + (size_t)(GROUPS + group) * (BT * H_); // t odd

    // ---- load our weight slice into LDS (once): Wlds[k][g*64+cc]
    for (int it = tid; it < KZ * 32; it += THREADS) {   // 32 chunks of 8 halves per row
        int k  = it >> 5;
        int np = (it & 31) * 8;       // n' start
        int g  = np >> 6;
        int cc = np & 63;
        *(v8h*)&Wlds[k][np] = *(const v8h*)(Wz + (size_t)k * NG + g * H_ + q * 64 + cc);
    }
    // ---- zero h region of Zs
    {
        int r = tid >> 3;
        int c = F_ + ((tid & 7) << 5);
        int4 z4 = make_int4(0, 0, 0, 0);
        *(int4*)&Zs[r][c]      = z4;
        *(int4*)&Zs[r][c + 8]  = z4;
        *(int4*)&Zs[r][c + 16] = z4;
        *(int4*)&Zs[r][c + 24] = z4;
    }
    // ---- stage x_0
    {
        int r  = tid >> 3;
        int c0 = (tid & 7) << 4;
        const float4* xp = (const float4*)(x + (((size_t)(batch0 + r)) * T_ + 0) * F_ + c0);
        float4 f0 = xp[0], f1 = xp[1], f2 = xp[2], f3 = xp[3];
        v8h h0, h1;
        h0[0]=(_Float16)f0.x; h0[1]=(_Float16)f0.y; h0[2]=(_Float16)f0.z; h0[3]=(_Float16)f0.w;
        h0[4]=(_Float16)f1.x; h0[5]=(_Float16)f1.y; h0[6]=(_Float16)f1.z; h0[7]=(_Float16)f1.w;
        h1[0]=(_Float16)f2.x; h1[1]=(_Float16)f2.y; h1[2]=(_Float16)f2.z; h1[3]=(_Float16)f2.w;
        h1[4]=(_Float16)f3.x; h1[5]=(_Float16)f3.y; h1[6]=(_Float16)f3.z; h1[7]=(_Float16)f3.w;
        *(v8h*)&Zs[r][c0]     = h0;
        *(v8h*)&Zs[r][c0 + 8] = h1;
    }

    // ---- biases for this wave's unit (global col = g*256 + q*64 + hsub*16 + rowA)
    const int hloc = q * 64 + hsub * 16 + rowA;   // our global hidden col
    const float bF = bias[0 * H_ + hloc];
    const float bU = bias[1 * H_ + hloc];
    const float bG = bias[2 * H_ + hloc];
    const float bO = bias[3 * H_ + hloc];

    float c_st[8];
#pragma unroll
    for (int i = 0; i < 8; ++i) c_st[i] = 0.f;

    __syncthreads();

    for (int t = 0; t < T_; ++t) {
        // opaque zero offset: defeats LICM so B-frags stay as in-loop ds_loads
        int woff = 0;
        asm volatile("" : "+v"(woff));
        const _Float16* wl = &Wlds[0][0] + woff;

        v8f aF = {bF,bF,bF,bF,bF,bF,bF,bF};
        v8f aU = {bU,bU,bU,bU,bU,bU,bU,bU};
        v8f aG = {bG,bG,bG,bG,bG,bG,bG,bG};
        v8f aO = {bO,bO,bO,bO,bO,bO,bO,bO};

#pragma unroll 2
        for (int kt = 0; kt < 12; ++kt) {
            const _Float16* zr = &Zs[mtile * 16 + rowA][kt * 32 + khalf * 8];
            v8h alo = *(const v8h*)zr;
            v8h ahi = *(const v8h*)(zr + 16);
            v16h a;
            ((v8h*)&a)[0] = alo;
            ((v8h*)&a)[1] = ahi;
            const _Float16* wr = wl + (kt * 32 + khalf * 8) * WP + hsub * 16 + rowA;
            v16h b0, b1, b2, b3;
#pragma unroll
            for (int s = 0; s < 16; ++s) {
                int kk = (s < 8) ? s : s + 8;
                b0[s] = wr[kk * WP + 0 * 64];
                b1[s] = wr[kk * WP + 1 * 64];
                b2[s] = wr[kk * WP + 2 * 64];
                b3[s] = wr[kk * WP + 3 * 64];
            }
            aF = wmma_f16(a, b0, aF);
            aU = wmma_f16(a, b1, aU);
            aG = wmma_f16(a, b2, aG);
            aO = wmma_f16(a, b3, aO);
        }

        // gate nonlinearity + cell update in registers; h_new -> global hbuf
        _Float16* hb = (t & 1) ? hb1 : hb0;
        float hn[8];
#pragma unroll
        for (int i = 0; i < 8; ++i) {
            float f  = sigf(aF[i]);
            float uu = sigf(aU[i]);
            float g  = tanhf_fast(aG[i]);
            float o  = sigf(aO[i]);
            float cn = f * c_st[i] + uu * g;
            c_st[i]  = cn;
            hn[i]    = o * tanhf_fast(cn);
        }
#pragma unroll
        for (int i = 0; i < 8; ++i) {
            int m = mtile * 16 + khalf * 8 + i;
            hb[m * H_ + hloc] = (_Float16)hn[i];
        }

        __threadfence();              // make our h slice visible (release)
        __syncthreads();              // whole WG done writing
        if (tid == 0) atomicAdd(cnt_g, 1);

        // overlap: stage x_{t+1} while other WGs arrive
        if (t + 1 < T_) {
            int r  = tid >> 3;
            int c0 = (tid & 7) << 4;
            const float4* xp = (const float4*)(x + (((size_t)(batch0 + r)) * T_ + (t + 1)) * F_ + c0);
            float4 f0 = xp[0], f1 = xp[1], f2 = xp[2], f3 = xp[3];
            v8h h0, h1;
            h0[0]=(_Float16)f0.x; h0[1]=(_Float16)f0.y; h0[2]=(_Float16)f0.z; h0[3]=(_Float16)f0.w;
            h0[4]=(_Float16)f1.x; h0[5]=(_Float16)f1.y; h0[6]=(_Float16)f1.z; h0[7]=(_Float16)f1.w;
            h1[0]=(_Float16)f2.x; h1[1]=(_Float16)f2.y; h1[2]=(_Float16)f2.z; h1[3]=(_Float16)f2.w;
            h1[4]=(_Float16)f3.x; h1[5]=(_Float16)f3.y; h1[6]=(_Float16)f3.z; h1[7]=(_Float16)f3.w;
            *(v8h*)&Zs[r][c0]     = h0;
            *(v8h*)&Zs[r][c0 + 8] = h1;
        }

        // group barrier: monotonic counter (zeroed each launch by prep kernel)
        if (tid == 0) {
            const int target = QWG * (t + 1);
            while (__hip_atomic_load(cnt_g, __ATOMIC_ACQUIRE, __HIP_MEMORY_SCOPE_AGENT) < target) {
                __builtin_amdgcn_s_sleep(1);
            }
        }
        __syncthreads();
        __threadfence();              // acquire: invalidate stale cached h

        // read full h_{t+1} (all 4 quarters) into Zs h-region
        {
            int r  = tid >> 3;
            int c0 = (tid & 7) << 5;  // 32 halves (64 B) per thread
#if __has_builtin(__builtin_amdgcn_global_load_async_to_lds_b128)
            as1_v4i* src = (as1_v4i*)(hb + (size_t)r * H_ + c0);
            as3_v4i* dst = (as3_v4i*)&Zs[r][F_ + c0];
            __builtin_amdgcn_global_load_async_to_lds_b128(src, dst, 0, 0);
            __builtin_amdgcn_global_load_async_to_lds_b128(src + 1, dst + 1, 0, 0);
            __builtin_amdgcn_global_load_async_to_lds_b128(src + 2, dst + 2, 0, 0);
            __builtin_amdgcn_global_load_async_to_lds_b128(src + 3, dst + 3, 0, 0);
#if __has_builtin(__builtin_amdgcn_s_wait_asynccnt)
            __builtin_amdgcn_s_wait_asynccnt(0);
#else
            asm volatile("s_wait_asynccnt 0x0" ::: "memory");
#endif
#else
            const v8h* src = (const v8h*)(hb + (size_t)r * H_ + c0);
            *(v8h*)&Zs[r][F_ + c0]      = src[0];
            *(v8h*)&Zs[r][F_ + c0 + 8]  = src[1];
            *(v8h*)&Zs[r][F_ + c0 + 16] = src[2];
            *(v8h*)&Zs[r][F_ + c0 + 24] = src[3];
#endif
        }
        __syncthreads();              // Zs ready for step t+1
    }

    // folded head from final h (f16 in Zs): out[b] = h.v + s   (quarter 0 only)
    if (q == 0 && tid < BT) {
        float acc = *shead;
        for (int j = 0; j < H_; ++j) acc += (float)Zs[tid][F_ + j] * vhead[j];
        out[batch0 + tid] = acc;
    }
}

// ---------------------------------------------------------------------------
extern "C" void kernel_launch(void* const* d_in, const int* in_sizes, int n_in,
                              void* d_out, int out_size, void* d_ws, size_t ws_size,
                              hipStream_t stream) {
    const float* x   = (const float*)d_in[0];
    const float* Wf  = (const float*)d_in[1];
    const float* bf  = (const float*)d_in[2];
    const float* Wu  = (const float*)d_in[3];
    const float* bu  = (const float*)d_in[4];
    const float* Wc  = (const float*)d_in[5];
    const float* bc  = (const float*)d_in[6];
    const float* Wo  = (const float*)d_in[7];
    const float* bo  = (const float*)d_in[8];
    const float* Wd1 = (const float*)d_in[9];
    const float* bd1 = (const float*)d_in[10];
    const float* Wd2 = (const float*)d_in[11];
    const float* bd2 = (const float*)d_in[12];

    char* ws = (char*)d_ws;
    _Float16* Wz   = (_Float16*)(ws + WZ_OFF);
    float* bias    = (float*)(ws + BIAS_OFF);
    float* vhead   = (float*)(ws + V_OFF);
    float* shead   = (float*)(ws + S_OFF);
    int* cnt       = (int*)(ws + CNT_OFF);
    _Float16* hbuf = (_Float16*)(ws + HBUF_OFF);
    float* out     = (float*)d_out;

    prep_weights<<<(KZ * NG + 255) / 256, 256, 0, stream>>>(Wf, bf, Wu, bu, Wc, bc, Wo, bo,
                                                            Wz, bias, cnt);
    prep_head<<<1, 256, 0, stream>>>(Wd1, bd1, Wd2, bd2, vhead, shead);
    lstm_main<<<GROUPS * QWG, THREADS, 0, stream>>>(x, Wz, bias, vhead, shead, hbuf, cnt, out);
}
